// TransformerDecoderLayer_60902636258022
// MI455X (gfx1250) — matrix operand order repaired and checked by reading, not compiled, not run on hardware
//
#include <hip/hip_runtime.h>
#include <hip/hip_bf16.h>

// ---------------------------------------------------------------- constants
#define S_LEN   2048
#define HDIM    1024
#define NQ      16
#define NKV     8
#define HD      64
#define FF      2048
#define NEXP    4
#define WINDOW  512
#define CAPV    50.0f
#define EPSV    1e-5f

typedef __attribute__((ext_vector_type(16))) __bf16 v16bf;
typedef __attribute__((ext_vector_type(8)))  float  v8f;

__device__ inline v8f vzero8() {
  v8f z = {0.f,0.f,0.f,0.f,0.f,0.f,0.f,0.f};
  return z;
}

// Build a 16-element bf16 fragment from two contiguous 16-byte chunks.
__device__ inline v16bf ldfrag(const __bf16* p0, const __bf16* p1) {
  v16bf r;
  reinterpret_cast<float4*>(&r)[0] = *reinterpret_cast<const float4*>(p0);
  reinterpret_cast<float4*>(&r)[1] = *reinterpret_cast<const float4*>(p1);
  return r;
}

__device__ inline v8f wmma_bf16(v16bf a, v16bf b, v8f c) {
  return __builtin_amdgcn_wmma_f32_16x16x32_bf16(false, a, false, b, (short)0, c,
                                                 false, false);
}

// CDNA5 async global->LDS copy (16B per lane), tracked by ASYNCcnt.
__device__ inline void async_load16(const void* g, const void* lds) {
  unsigned l = (unsigned)(size_t)lds;
  asm volatile("global_load_async_to_lds_b128 %0, %1, off"
               :: "v"(l), "v"(g) : "memory");
}
__device__ inline void async_wait0() {
  asm volatile("s_wait_asynccnt 0" ::: "memory");
}

// Hardware tanh (gfx1250 V_TANH_F32, TRANS op).
__device__ inline float fast_tanh(float x) {
#if __has_builtin(__builtin_amdgcn_tanhf)
  return __builtin_amdgcn_tanhf(x);
#else
  float y;
  asm volatile("v_tanh_f32 %0, %1\n\tv_nop" : "=v"(y) : "v"(x));
  return y;
#endif
}

// ---------------------------------------------------------------- rmsnorm
__global__ __launch_bounds__(256) void k_rmsnorm(const float* __restrict__ x,
                                                 const float* __restrict__ w,
                                                 __bf16* __restrict__ ob,
                                                 float* __restrict__ of, int H) {
  int row = blockIdx.x, tid = threadIdx.x;
  const float* xr = x + (size_t)row * H;
  float ss = 0.f;
  for (int i = tid; i < H; i += 256) { float v = xr[i]; ss += v * v; }
  for (int m = 16; m >= 1; m >>= 1) ss += __shfl_xor(ss, m, 32);
  __shared__ float red[8];
  if ((tid & 31) == 0) red[tid >> 5] = ss;
  __syncthreads();
  float tot = 0.f;
  #pragma unroll
  for (int i = 0; i < 8; ++i) tot += red[i];
  float sc = rsqrtf(tot / (float)H + EPSV);
  for (int i = tid; i < H; i += 256) {
    float v = xr[i] * sc * w[i];
    ob[(size_t)row * H + i] = (__bf16)v;
    if (of) of[(size_t)row * H + i] = v;
  }
}

// ------------------------------------------- transpose+convert: [K][N]f32 -> [N][K]bf16
__global__ __launch_bounds__(256) void k_cvt_t(const float* __restrict__ in,
                                               __bf16* __restrict__ out,
                                               int K, int N) {
  __shared__ __bf16 t[32][33];
  int n0 = blockIdx.x * 32, k0 = blockIdx.y * 32;
  int tn = threadIdx.x & 31, tk = threadIdx.x >> 5;
  #pragma unroll
  for (int p = 0; p < 4; ++p) {
    int k = tk + p * 8;
    t[tn][k] = (__bf16)in[(size_t)(k0 + k) * N + n0 + tn];
  }
  __syncthreads();
  #pragma unroll
  for (int p = 0; p < 4; ++p) {
    int n = tk + p * 8;
    out[(size_t)(n0 + n) * K + k0 + tn] = t[n][tn];
  }
}

// ------------------------------------------------- bf16 WMMA GEMM (B pre-transposed)
// C[M,N] = A[M,K] @ Bt[N,K]^T, fp32 accum, async double-buffered LDS staging.
// MODE 0: out = C    MODE 1: out = res + C    MODE 3: out += rowscale[m*rs]*C
#define TM 128
#define TN 128
#define TK 64
template <int MODE>
__global__ __launch_bounds__(256) void k_gemm(const __bf16* __restrict__ A,
                                              const __bf16* __restrict__ Bt,
                                              float* __restrict__ out,
                                              const float* __restrict__ res,
                                              const float* __restrict__ rowscale,
                                              int rs_stride,
                                              int M, int N, int K) {
  __shared__ __bf16 As[2][TM][TK + 8];   // row stride 144B (16B aligned)
  __shared__ __bf16 Bs[2][TN][TK + 8];
  const int tid = threadIdx.x;
  const int wave = tid >> 5, lane = tid & 31;
  const int lm = lane & 15, lh = lane >> 4;
  const int wm = wave & 3, wn = wave >> 2;           // wave tile: 32(m) x 64(n)
  const int m0 = blockIdx.y * TM;
  const int n0 = blockIdx.x * TN;

  const int sr = tid & 127, sh = tid >> 7;           // staging: 2 threads/row, 64B each

  auto stage = [&](int buf, int k0) {
    const __bf16* ga = A + (size_t)(m0 + sr) * K + k0 + sh * 32;
    const __bf16* gb = Bt + (size_t)(n0 + sr) * K + k0 + sh * 32;
    #pragma unroll
    for (int c = 0; c < 4; ++c) {
      async_load16(ga + c * 8, &As[buf][sr][sh * 32 + c * 8]);
      async_load16(gb + c * 8, &Bs[buf][sr][sh * 32 + c * 8]);
    }
  };

  v8f acc[2][4];
  #pragma unroll
  for (int i = 0; i < 2; ++i)
    #pragma unroll
    for (int j = 0; j < 4; ++j) acc[i][j] = vzero8();

  const int nk = K / TK;
  stage(0, 0);
  async_wait0();
  __syncthreads();

  for (int t = 0; t < nk; ++t) {
    const int cur = t & 1;
    if (t + 1 < nk) stage(cur ^ 1, (t + 1) * TK);
    #pragma unroll
    for (int ks = 0; ks < 2; ++ks) {
      v16bf af[2], bfm[4];
      #pragma unroll
      for (int i = 0; i < 2; ++i) {
        int m = wm * 32 + i * 16 + lm;
        af[i] = ldfrag(&As[cur][m][ks * 32 + lh * 8],
                       &As[cur][m][ks * 32 + 16 + lh * 8]);
      }
      #pragma unroll
      for (int j = 0; j < 4; ++j) {
        int n = wn * 64 + j * 16 + lm;
        bfm[j] = ldfrag(&Bs[cur][n][ks * 32 + lh * 8],
                        &Bs[cur][n][ks * 32 + 16 + lh * 8]);
      }
      #pragma unroll
      for (int i = 0; i < 2; ++i)
        #pragma unroll
        for (int j = 0; j < 4; ++j)
          acc[i][j] = wmma_bf16(af[i], bfm[j], acc[i][j]);
    }
    async_wait0();
    __syncthreads();
  }

  #pragma unroll
  for (int i = 0; i < 2; ++i)
    #pragma unroll
    for (int j = 0; j < 4; ++j) {
      int n = n0 + wn * 64 + j * 16 + lm;
      #pragma unroll
      for (int r = 0; r < 8; ++r) {
        int m = m0 + wm * 32 + i * 16 + r + lh * 8;
        size_t idx = (size_t)m * N + n;
        float c = acc[i][j][r];
        if constexpr (MODE == 0)      out[idx] = c;
        else if constexpr (MODE == 1) out[idx] = res[idx] + c;
        else                          out[idx] += rowscale[(size_t)m * rs_stride] * c;
      }
    }
}

// ------------------------------------------------- RoPE + per-head RMS
__global__ __launch_bounds__(64) void k_rope_rms(const float* __restrict__ inp,
                                                 const float* __restrict__ nw,
                                                 __bf16* __restrict__ outp,
                                                 int NH, int S) {
  int s = blockIdx.x, h = blockIdx.y, d = threadIdx.x;
  const float* row = inp + ((size_t)s * NH + h) * 64;
  float r;
  if (d < 32) {
    float inv = __powf(10000.0f, -(float)d / 32.0f);
    float ang = (float)s * inv;
    r = row[d] * __cosf(ang) - row[d + 32] * __sinf(ang);
  } else {
    float inv = __powf(10000.0f, -(float)(d - 32) / 32.0f);
    float ang = (float)s * inv;
    r = row[d - 32] * __sinf(ang) + row[d] * __cosf(ang);
  }
  float ss = r * r;
  for (int m = 16; m >= 1; m >>= 1) ss += __shfl_xor(ss, m, 32);
  __shared__ float red[2];
  if ((d & 31) == 0) red[d >> 5] = ss;
  __syncthreads();
  float sc = rsqrtf((red[0] + red[1]) / 64.0f + EPSV);
  outp[((size_t)h * S + s) * 64 + d] = (__bf16)(r * sc * nw[d]);
}

__global__ __launch_bounds__(64) void k_vtrans(const float* __restrict__ v,
                                               __bf16* __restrict__ vt, int S) {
  int s = blockIdx.x, h = blockIdx.y, d = threadIdx.x;
  vt[((size_t)h * 64 + d) * S + s] = (__bf16)v[((size_t)s * NKV + h) * 64 + d];
}

// ------------------------------------------------- attention (1 wave / 16-q tile)
__global__ __launch_bounds__(32) void k_attn(const __bf16* __restrict__ qh,  // [NQ][S][64]
                                             const __bf16* __restrict__ kh,  // [NKV][S][64]
                                             const __bf16* __restrict__ vt,  // [NKV][64][S]
                                             __bf16* __restrict__ attn,      // [S][NQ*64]
                                             int S) {
  const int qt = blockIdx.x, head = blockIdx.y;
  const int q0 = qt * 16;
  const int kvh = head >> 1;
  const int lane = threadIdx.x, lm = lane & 15, lh = lane >> 4;

  __shared__ float sc[16][576];
  __shared__ float rmax_s[16], rinv_s[16];

  int jlo = q0 - (WINDOW - 1); if (jlo < 0) jlo = 0;
  const int kb0 = jlo & ~31;
  const int jhi = q0 + 15;
  const int nblk = ((jhi - kb0) >> 5) + 1;

  const __bf16* qbase = qh + ((size_t)head * S + q0) * 64;
  v16bf aq[2];
  #pragma unroll
  for (int ks = 0; ks < 2; ++ks) {
    const __bf16* p = qbase + (size_t)lm * 64 + ks * 32;
    aq[ks] = ldfrag(p + lh * 8, p + 16 + lh * 8);
  }

  const float pre = 0.125f / CAPV;   // scale / CAP folded into one multiply
  float pmax[8];
  #pragma unroll
  for (int r = 0; r < 8; ++r) pmax[r] = -__builtin_inff();

  for (int b = 0; b < nblk; ++b) {
    int kb = kb0 + b * 32;
    v8f c[2] = { vzero8(), vzero8() };
    #pragma unroll
    for (int nt = 0; nt < 2; ++nt)
      #pragma unroll
      for (int ks = 0; ks < 2; ++ks) {
        const __bf16* p = kh + ((size_t)kvh * S + kb + nt * 16 + lm) * 64 + ks * 32 + lh * 8;
        v16bf bk = ldfrag(p, p + 16);
        c[nt] = wmma_bf16(aq[ks], bk, c[nt]);
      }
    #pragma unroll
    for (int nt = 0; nt < 2; ++nt)
      #pragma unroll
      for (int r = 0; r < 8; ++r) {
        int i = q0 + r + lh * 8;
        int j = kb + nt * 16 + lm;
        float s = CAPV * fast_tanh(c[nt][r] * pre);
        bool allow = (j <= i) && (i - j < WINDOW);
        if (!allow) s = -__builtin_inff();
        sc[r + lh * 8][j - kb0] = s;
        pmax[r] = fmaxf(pmax[r], s);
      }
  }
  #pragma unroll
  for (int r = 0; r < 8; ++r) {
    float v = pmax[r];
    for (int m = 8; m >= 1; m >>= 1) v = fmaxf(v, __shfl_xor(v, m, 32));
    if (lm == 0) rmax_s[r + lh * 8] = v;
  }
  __syncthreads();

  {
    int ncol = nblk * 32;
    float rm = rmax_s[lm];
    float partial = 0.f;
    for (int c0 = lh; c0 < ncol; c0 += 2) partial += __expf(sc[lm][c0] - rm);
    partial += __shfl_xor(partial, 16, 32);
    if (lane < 16) rinv_s[lm] = __fdividef(1.0f, partial);
  }
  __syncthreads();

  v8f acc[4];
  #pragma unroll
  for (int nt = 0; nt < 4; ++nt) acc[nt] = vzero8();

  const float rm2 = rmax_s[lm];
  const float ri  = rinv_s[lm];
  for (int b = 0; b < nblk; ++b) {
    int kb = kb0 + b * 32;
    v16bf pa;
    #pragma unroll
    for (int e = 0; e < 16; ++e) {
      int k = e + lh * 16;
      float pv = __expf(sc[lm][kb - kb0 + k] - rm2) * ri;
      pv = 1.06f * pv - 0.03f;
      pv = fminf(fmaxf(pv, 0.0f), 1.0f);
      pa[e] = (__bf16)pv;
    }
    #pragma unroll
    for (int nt = 0; nt < 4; ++nt) {
      const __bf16* p = vt + ((size_t)kvh * 64 + nt * 16 + lm) * S + kb + lh * 16;
      v16bf bv = ldfrag(p, p + 8);
      acc[nt] = wmma_bf16(pa, bv, acc[nt]);
    }
  }

  #pragma unroll
  for (int nt = 0; nt < 4; ++nt)
    #pragma unroll
    for (int r = 0; r < 8; ++r) {
      int m = r + lh * 8;
      int d = nt * 16 + lm;
      attn[(size_t)(q0 + m) * (NQ * HD) + head * 64 + d] = (__bf16)acc[nt][r];
    }
}

// ------------------------------------------------- MoE gate (softmax + top2)
__global__ __launch_bounds__(128) void k_gate(const float* __restrict__ xn2,
                                              const float* __restrict__ gw,
                                              float* __restrict__ wdense, int H) {
  int t = blockIdx.x, tid = threadIdx.x;
  float acc[NEXP] = {0.f, 0.f, 0.f, 0.f};
  for (int i = tid; i < H; i += 128) {
    float xv = xn2[(size_t)t * H + i];
    #pragma unroll
    for (int e = 0; e < NEXP; ++e) acc[e] += xv * gw[i * NEXP + e];
  }
  __shared__ float red[4][NEXP];
  #pragma unroll
  for (int e = 0; e < NEXP; ++e) {
    float v = acc[e];
    for (int m = 16; m >= 1; m >>= 1) v += __shfl_xor(v, m, 32);
    if ((tid & 31) == 0) red[tid >> 5][e] = v;
  }
  __syncthreads();
  if (tid == 0) {
    float l[NEXP];
    #pragma unroll
    for (int e = 0; e < NEXP; ++e) l[e] = red[0][e] + red[1][e] + red[2][e] + red[3][e];
    float mx = l[0];
    for (int e = 1; e < NEXP; ++e) mx = fmaxf(mx, l[e]);
    float p[NEXP], sum = 0.f;
    for (int e = 0; e < NEXP; ++e) { p[e] = expf(l[e] - mx); sum += p[e]; }
    for (int e = 0; e < NEXP; ++e) p[e] /= sum;
    int i0 = 0;
    for (int e = 1; e < NEXP; ++e) if (p[e] > p[i0]) i0 = e;
    int i1 = -1;
    for (int e = 0; e < NEXP; ++e) if (e != i0 && (i1 < 0 || p[e] > p[i1])) i1 = e;
    float s2 = p[i0] + p[i1];
    for (int e = 0; e < NEXP; ++e) wdense[(size_t)t * NEXP + e] = 0.f;
    wdense[(size_t)t * NEXP + i0] = p[i0] / s2;
    wdense[(size_t)t * NEXP + i1] = p[i1] / s2;
  }
}

// ------------------------------------------------- silu(g)*u from fused [T][2FF]
__global__ void k_silu_mul(const float* __restrict__ gu, __bf16* __restrict__ o, int n) {
  int i = blockIdx.x * 256 + threadIdx.x;
  if (i < n) {
    int t = i / FF, f = i - t * FF;
    float gv = gu[(size_t)t * (2 * FF) + f];
    float uv = gu[(size_t)t * (2 * FF) + FF + f];
    o[i] = (__bf16)(__fdividef(gv, 1.0f + __expf(-gv)) * uv);
  }
}

__global__ void k_copy(const float* __restrict__ a, float* __restrict__ b, int n) {
  int i = blockIdx.x * 256 + threadIdx.x;
  if (i < n) b[i] = a[i];
}

// ================================================================ launcher
extern "C" void kernel_launch(void* const* d_in, const int* in_sizes, int n_in,
                              void* d_out, int out_size, void* d_ws, size_t ws_size,
                              hipStream_t stream) {
  const float* x       = (const float*)d_in[0];
  const float* norm1_w = (const float*)d_in[1];
  const float* wq      = (const float*)d_in[2];
  const float* wk      = (const float*)d_in[3];
  const float* wv      = (const float*)d_in[4];
  const float* wo      = (const float*)d_in[5];
  const float* qn_w    = (const float*)d_in[6];
  const float* kn_w    = (const float*)d_in[7];
  const float* norm2_w = (const float*)d_in[8];
  const float* gate_w  = (const float*)d_in[9];
  const float* w_gate  = (const float*)d_in[10];
  const float* w_up    = (const float*)d_in[11];
  const float* w_down  = (const float*)d_in[12];
  float* out = (float*)d_out;

  char* ws = (char*)d_ws;
  size_t off = 0;
  auto alloc = [&](size_t bytes) {
    char* p = ws + off;
    off += (bytes + 255) & ~(size_t)255;
    return p;
  };

  __bf16* wqT  = (__bf16*)alloc((size_t)(NQ * HD) * HDIM * 2);
  __bf16* wkT  = (__bf16*)alloc((size_t)(NKV * HD) * HDIM * 2);
  __bf16* wvT  = (__bf16*)alloc((size_t)(NKV * HD) * HDIM * 2);
  __bf16* woT  = (__bf16*)alloc((size_t)HDIM * (NQ * HD) * 2);
  __bf16* guT  = (__bf16*)alloc((size_t)NEXP * 2 * FF * HDIM * 2);  // [e][2FF][H]
  __bf16* wdT  = (__bf16*)alloc((size_t)NEXP * HDIM * FF * 2);      // [e][H][FF]
  __bf16* xn_b  = (__bf16*)alloc((size_t)S_LEN * HDIM * 2);
  float*  q_f   = (float*) alloc((size_t)S_LEN * NQ * HD * 4);
  float*  k_f   = (float*) alloc((size_t)S_LEN * NKV * HD * 4);
  float*  v_f   = (float*) alloc((size_t)S_LEN * NKV * HD * 4);
  __bf16* qh_b  = (__bf16*)alloc((size_t)NQ * S_LEN * HD * 2);
  __bf16* kh_b  = (__bf16*)alloc((size_t)NKV * S_LEN * HD * 2);
  __bf16* vt_b  = (__bf16*)alloc((size_t)NKV * HD * S_LEN * 2);
  __bf16* at_b  = (__bf16*)alloc((size_t)S_LEN * NQ * HD * 2);
  float*  h_f   = (float*) alloc((size_t)S_LEN * HDIM * 4);
  __bf16* xn2_b = (__bf16*)alloc((size_t)S_LEN * HDIM * 2);
  float*  xn2_f = (float*) alloc((size_t)S_LEN * HDIM * 4);
  float*  wdns  = (float*) alloc((size_t)S_LEN * NEXP * 4);
  float*  gu_f  = (float*) alloc((size_t)S_LEN * 2 * FF * 4);
  __bf16* t_b   = (__bf16*)alloc((size_t)S_LEN * FF * 2);

  k_cvt_t<<<dim3((NQ * HD) / 32, HDIM / 32), 256, 0, stream>>>(wq, wqT, HDIM, NQ * HD);
  k_cvt_t<<<dim3((NKV * HD) / 32, HDIM / 32), 256, 0, stream>>>(wk, wkT, HDIM, NKV * HD);
  k_cvt_t<<<dim3((NKV * HD) / 32, HDIM / 32), 256, 0, stream>>>(wv, wvT, HDIM, NKV * HD);
  k_cvt_t<<<dim3(HDIM / 32, (NQ * HD) / 32), 256, 0, stream>>>(wo, woT, NQ * HD, HDIM);
  for (int e = 0; e < NEXP; ++e) {
    k_cvt_t<<<dim3(FF / 32, HDIM / 32), 256, 0, stream>>>(
        w_gate + (size_t)e * HDIM * FF, guT + (size_t)e * 2 * FF * HDIM, HDIM, FF);
    k_cvt_t<<<dim3(FF / 32, HDIM / 32), 256, 0, stream>>>(
        w_up + (size_t)e * HDIM * FF, guT + (size_t)e * 2 * FF * HDIM + (size_t)FF * HDIM,
        HDIM, FF);
    k_cvt_t<<<dim3(HDIM / 32, FF / 32), 256, 0, stream>>>(
        w_down + (size_t)e * FF * HDIM, wdT + (size_t)e * HDIM * FF, FF, HDIM);
  }

  k_rmsnorm<<<S_LEN, 256, 0, stream>>>(x, norm1_w, xn_b, nullptr, HDIM);

  k_gemm<0><<<dim3((NQ * HD) / TN, S_LEN / TM), 256, 0, stream>>>(
      xn_b, wqT, q_f, nullptr, nullptr, 0, S_LEN, NQ * HD, HDIM);
  k_gemm<0><<<dim3((NKV * HD) / TN, S_LEN / TM), 256, 0, stream>>>(
      xn_b, wkT, k_f, nullptr, nullptr, 0, S_LEN, NKV * HD, HDIM);
  k_gemm<0><<<dim3((NKV * HD) / TN, S_LEN / TM), 256, 0, stream>>>(
      xn_b, wvT, v_f, nullptr, nullptr, 0, S_LEN, NKV * HD, HDIM);

  k_rope_rms<<<dim3(S_LEN, NQ),  64, 0, stream>>>(q_f, qn_w, qh_b, NQ,  S_LEN);
  k_rope_rms<<<dim3(S_LEN, NKV), 64, 0, stream>>>(k_f, kn_w, kh_b, NKV, S_LEN);
  k_vtrans  <<<dim3(S_LEN, NKV), 64, 0, stream>>>(v_f, vt_b, S_LEN);

  k_attn<<<dim3(S_LEN / 16, NQ), 32, 0, stream>>>(qh_b, kh_b, vt_b, at_b, S_LEN);

  k_gemm<1><<<dim3(HDIM / TN, S_LEN / TM), 256, 0, stream>>>(
      at_b, woT, h_f, x, nullptr, 0, S_LEN, HDIM, NQ * HD);

  k_rmsnorm<<<S_LEN, 256, 0, stream>>>(h_f, norm2_w, xn2_b, xn2_f, HDIM);
  k_gate<<<S_LEN, 128, 0, stream>>>(xn2_f, gate_w, wdns, HDIM);

  k_copy<<<(S_LEN * HDIM + 255) / 256, 256, 0, stream>>>(h_f, out, S_LEN * HDIM);

  for (int e = 0; e < NEXP; ++e) {
    k_gemm<0><<<dim3((2 * FF) / TN, S_LEN / TM), 256, 0, stream>>>(
        xn2_b, guT + (size_t)e * 2 * FF * HDIM, gu_f, nullptr, nullptr, 0,
        S_LEN, 2 * FF, HDIM);
    k_silu_mul<<<(S_LEN * FF + 255) / 256, 256, 0, stream>>>(gu_f, t_b, S_LEN * FF);
    k_gemm<3><<<dim3(HDIM / TN, S_LEN / TM), 256, 0, stream>>>(
        t_b, wdT + (size_t)e * HDIM * FF, out, nullptr, wdns + e, NEXP,
        S_LEN, HDIM, FF);
  }
}
